// SelfAttention_14757507629919
// MI455X (gfx1250) — matrix operand order repaired and checked by reading, not compiled, not run on hardware
//
#include <hip/hip_runtime.h>
#include <hip/hip_bf16.h>

typedef __attribute__((ext_vector_type(16))) _Float16 v16h;
typedef __attribute__((ext_vector_type(8)))  _Float16 v8h;
typedef __attribute__((ext_vector_type(4)))  _Float16 v4h;
typedef __attribute__((ext_vector_type(8)))  float    v8f;
typedef __attribute__((ext_vector_type(4)))  float    v4f;

enum : int { Bn = 2, Tn = 2048, Cn = 1024, Hn = 16, Dn = 64 };

static __device__ __forceinline__ v8f zero8() {
  v8f z;
#pragma unroll
  for (int i = 0; i < 8; ++i) z[i] = 0.0f;
  return z;
}

static __device__ __forceinline__ v8f wmma16(v16h a, v16h b, v8f c) {
  // D = A(16x32 f16) x B(32x16 f16) + C(16x16 f32)
  return __builtin_amdgcn_wmma_f32_16x16x32_f16(false, a, false, b, (short)0, c,
                                                false, false);
}

// A-fragment: 16x32 (MxK) f16 from row-major tile p[row][col], ld elements.
// lanes 0-15: row=lane, elems 0..7 = K k0..k0+7, elems 8..15 = K k0+16..k0+23;
// lanes 16-31: row=lane-16, K shifted by +8 / +24.
static __device__ __forceinline__ v16h load_afrag(const _Float16* p, int ld, int k0) {
  int lane = threadIdx.x & 31;
  int half = lane >> 4;
  int row  = lane & 15;
  const _Float16* q = p + (size_t)row * ld + k0 + half * 8;
  union { v16h v; v8h h[2]; } u;
  u.h[0] = *(const v8h*)(q);
  u.h[1] = *(const v8h*)(q + 16);
  return u.v;
}

// B-fragment: 32x16 (KxN) f16 where element (k,n) = p[n*ld + k]  (p points at row n0).
// lanes 0-15: col=lane, elems 0..7 = K k0..k0+7, elems 8..15 = K k0+16..k0+23;
// lanes 16-31: col=lane-16, K shifted by +8 / +24.
static __device__ __forceinline__ v16h load_bfrag(const _Float16* p, int ld, int k0) {
  int lane = threadIdx.x & 31;
  int half = lane >> 4;
  int col  = lane & 15;
  const _Float16* q = p + (size_t)col * ld + k0 + half * 8;
  union { v16h v; v8h h[2]; } u;
  u.h[0] = *(const v8h*)(q);
  u.h[1] = *(const v8h*)(q + 16);
  return u.v;
}

// ---------------------------------------------------------------- converts (x4 vectorized)
__global__ void __launch_bounds__(256) cvt_f32_f16_kernel(const float* __restrict__ s,
                                                          _Float16* __restrict__ d, int n4) {
  int i = blockIdx.x * blockDim.x + threadIdx.x;
  int stride = gridDim.x * blockDim.x;
  for (; i < n4; i += stride) {
    v4f x = ((const v4f*)s)[i];
    v4h y;
#pragma unroll
    for (int j = 0; j < 4; ++j) y[j] = (_Float16)x[j];
    ((v4h*)d)[i] = y;
  }
}

// ---------------------------------------------------------------- fused QKV GEMM
// blockIdx.y selects {Q,K,V}. out = X(4096x1024) @ W.T + bias, split-head:
//   Q,K : out[b][h][t][d]   ;   V : out[b][h][d][t] (transposed, packed u32 stores)
// Wave tile 64x64: 4 A-frags x 4 B-frags -> 16 WMMAs per K-step.
__global__ void __launch_bounds__(256) qkv_gemm_kernel(
    const _Float16* __restrict__ X,
    const _Float16* __restrict__ Wq, const float* __restrict__ bq, _Float16* __restrict__ Qo,
    const _Float16* __restrict__ Wk, const float* __restrict__ bk, _Float16* __restrict__ Ko,
    const _Float16* __restrict__ Wv, const float* __restrict__ bv, _Float16* __restrict__ Vo) {
  const _Float16* W;
  const float* bias;
  _Float16* out;
  int vmode = 0;
  if (blockIdx.y == 0)      { W = Wq; bias = bq; out = Qo; }
  else if (blockIdx.y == 1) { W = Wk; bias = bk; out = Ko; }
  else                      { W = Wv; bias = bv; out = Vo; vmode = 1; }

  int wave  = threadIdx.x >> 5;
  int wid   = blockIdx.x * (blockDim.x >> 5) + wave;  // 1024 waves
  int ntile = wid & 15;                               // 16 tiles of N=64
  int mtile = wid >> 4;                               // 64 tiles of M=64
  int m0 = mtile * 64;
  int n0 = ntile * 64;
  int lane = threadIdx.x & 31, half = lane >> 4, lan = lane & 15;

  v8f acc[4][4];
#pragma unroll
  for (int mi = 0; mi < 4; ++mi)
#pragma unroll
    for (int f = 0; f < 4; ++f) acc[mi][f] = zero8();

  for (int k0 = 0; k0 < Cn; k0 += 32) {
    v16h a[4];
#pragma unroll
    for (int mi = 0; mi < 4; ++mi)
      a[mi] = load_afrag(X + (size_t)(m0 + 16 * mi) * Cn, Cn, k0);
#pragma unroll
    for (int f = 0; f < 4; ++f) {
      v16h bf = load_bfrag(W + (size_t)(n0 + 16 * f) * Cn, Cn, k0);
#pragma unroll
      for (int mi = 0; mi < 4; ++mi) acc[mi][f] = wmma16(a[mi], bf, acc[mi][f]);
    }
  }

#pragma unroll
  for (int mi = 0; mi < 4; ++mi)
#pragma unroll
    for (int f = 0; f < 4; ++f) {
      int col = n0 + lan + 16 * f;
      float bvv = bias[col];
      int h = col >> 6, d = col & 63;
      if (vmode == 0) {
#pragma unroll
        for (int i = 0; i < 8; ++i) {
          int r = m0 + mi * 16 + i + 8 * half;        // global row in [0, B*T)
          int b = r >> 11, t = r & (Tn - 1);
          out[(((size_t)b * Hn + h) * Tn + t) * Dn + d] =
              (_Float16)(acc[mi][f][i] + bvv);
        }
      } else {
        // [b][h][d][t]: elements i,i+1 are t,t+1 -> pack into one u32 store
#pragma unroll
        for (int i = 0; i < 8; i += 2) {
          int r = m0 + mi * 16 + i + 8 * half;
          int b = r >> 11, t = r & (Tn - 1);
          union { _Float16 h2[2]; unsigned u; } pk;
          pk.h2[0] = (_Float16)(acc[mi][f][i] + bvv);
          pk.h2[1] = (_Float16)(acc[mi][f][i + 1] + bvv);
          *(unsigned*)&out[(((size_t)b * Hn + h) * Dn + d) * Tn + t] = pk.u;
        }
      }
    }
}

// ---------------------------------------------------------------- flash attention
// One wave handles 32 query rows of one (b,h). Online softmax over s in tiles of 32.
__global__ void __launch_bounds__(128) flash_attn_kernel(const _Float16* __restrict__ Qh,
                                                         const _Float16* __restrict__ Kh,
                                                         const _Float16* __restrict__ Vth,
                                                         _Float16* __restrict__ Yh) {
  __shared__ _Float16 pst[4][32][32];   // per-wave P staging (C-layout -> A-layout)
  int wave = threadIdx.x >> 5;
  int wid  = blockIdx.x * 4 + wave;
  int bh = wid >> 6;                    // 64 t-tiles per (b,h)
  int tt = wid & 63;
  int b = bh >> 4, h = bh & 15;
  int t0 = tt * 32;
  int lane = threadIdx.x & 31, half = lane >> 4, lan = lane & 15;

  const _Float16* Qp = Qh + ((size_t)bh * Tn + t0) * Dn;
  const _Float16* Kp = Kh + (size_t)bh * Tn * Dn;
  const _Float16* Vp = Vth + (size_t)bh * Dn * Tn;

  v16h aQ[2][2];
#pragma unroll
  for (int mi = 0; mi < 2; ++mi)
#pragma unroll
    for (int kk = 0; kk < 2; ++kk)
      aQ[mi][kk] = load_afrag(Qp + (size_t)mi * 16 * Dn, Dn, kk * 32);

  v8f O[2][4], mrow[2], lrow[2];
#pragma unroll
  for (int mi = 0; mi < 2; ++mi) {
#pragma unroll
    for (int f = 0; f < 4; ++f) O[mi][f] = zero8();
#pragma unroll
    for (int i = 0; i < 8; ++i) { mrow[mi][i] = -1e30f; lrow[mi][i] = 0.0f; }
  }
  const float scale = 0.125f;           // 1/sqrt(64)

  for (int s0 = 0; s0 < Tn; s0 += 32) {
    v16h bK[2][2];
#pragma unroll
    for (int si = 0; si < 2; ++si)
#pragma unroll
      for (int kk = 0; kk < 2; ++kk)
        bK[si][kk] = load_bfrag(Kp + (size_t)(s0 + si * 16) * Dn, Dn, kk * 32);

    v8f S[2][2];
#pragma unroll
    for (int mi = 0; mi < 2; ++mi)
#pragma unroll
      for (int si = 0; si < 2; ++si) {
        v8f z = zero8();
        z = wmma16(aQ[mi][0], bK[si][0], z);
        S[mi][si] = wmma16(aQ[mi][1], bK[si][1], z);
      }

#pragma unroll
    for (int mi = 0; mi < 2; ++mi) {
      v8f mx, p0, p1, rs, mn, al;
#pragma unroll
      for (int i = 0; i < 8; ++i) {
        S[mi][0][i] *= scale;
        S[mi][1][i] *= scale;
        mx[i] = fmaxf(S[mi][0][i], S[mi][1][i]);
      }
      // row max: a row lives in one 16-lane half; xor masks 1..8 stay in-half
#pragma unroll
      for (int off = 1; off < 16; off <<= 1)
#pragma unroll
        for (int i = 0; i < 8; ++i)
          mx[i] = fmaxf(mx[i], __shfl_xor(mx[i], off, 32));
#pragma unroll
      for (int i = 0; i < 8; ++i) {
        mn[i] = fmaxf(mrow[mi][i], mx[i]);
        al[i] = __expf(mrow[mi][i] - mn[i]);
        p0[i] = __expf(S[mi][0][i] - mn[i]);
        p1[i] = __expf(S[mi][1][i] - mn[i]);
        rs[i] = p0[i] + p1[i];
      }
#pragma unroll
      for (int off = 1; off < 16; off <<= 1)
#pragma unroll
        for (int i = 0; i < 8; ++i)
          rs[i] += __shfl_xor(rs[i], off, 32);
#pragma unroll
      for (int i = 0; i < 8; ++i) {
        lrow[mi][i] = lrow[mi][i] * al[i] + rs[i];
        mrow[mi][i] = mn[i];
      }
#pragma unroll
      for (int f = 0; f < 4; ++f)
#pragma unroll
        for (int i = 0; i < 8; ++i) O[mi][f][i] *= al[i];
      // stage P (C-layout: row = i+8*half, col = lan (+16 for second frag))
#pragma unroll
      for (int i = 0; i < 8; ++i) {
        pst[wave][mi * 16 + i + 8 * half][lan]      = (_Float16)p0[i];
        pst[wave][mi * 16 + i + 8 * half][lan + 16] = (_Float16)p1[i];
      }
    }
    // Per-wave staging only: LDS is in-order within a wave; force completion of
    // the ds_stores before the transposed ds_loads (no block barrier needed).
    asm volatile("s_wait_dscnt 0" ::: "memory");

    v16h aP0 = load_afrag(&pst[wave][0][0], 32, 0);
    v16h aP1 = load_afrag(&pst[wave][16][0], 32, 0);
#pragma unroll
    for (int f = 0; f < 4; ++f) {
      v16h bV = load_bfrag(Vp + (size_t)(f * 16) * Tn, Tn, s0);
      O[0][f] = wmma16(aP0, bV, O[0][f]);
      O[1][f] = wmma16(aP1, bV, O[1][f]);
    }
    // In-order LDS pipeline per wave: next iteration's stores cannot bypass
    // these loads, so no trailing barrier is required.
  }

#pragma unroll
  for (int mi = 0; mi < 2; ++mi) {
    v8f rec;
#pragma unroll
    for (int i = 0; i < 8; ++i) rec[i] = 1.0f / lrow[mi][i];
#pragma unroll
    for (int f = 0; f < 4; ++f)
#pragma unroll
      for (int i = 0; i < 8; ++i) {
        int t = t0 + mi * 16 + i + 8 * half;
        int c = h * Dn + lan + 16 * f;
        Yh[((size_t)b * Tn + t) * Cn + c] = (_Float16)(O[mi][f][i] * rec[i]);
      }
  }
}

// ---------------------------------------------------------------- output projection
// Wave tile 64x64, f32 output + bias.
__global__ void __launch_bounds__(256) proj_gemm_kernel(const _Float16* __restrict__ X,
                                                        const _Float16* __restrict__ W,
                                                        const float* __restrict__ bias,
                                                        float* __restrict__ out) {
  int wave  = threadIdx.x >> 5;
  int wid   = blockIdx.x * (blockDim.x >> 5) + wave;
  int ntile = wid & 15;
  int mtile = wid >> 4;
  int m0 = mtile * 64;
  int n0 = ntile * 64;
  int lane = threadIdx.x & 31, half = lane >> 4, lan = lane & 15;

  v8f acc[4][4];
#pragma unroll
  for (int mi = 0; mi < 4; ++mi)
#pragma unroll
    for (int f = 0; f < 4; ++f) acc[mi][f] = zero8();

  for (int k0 = 0; k0 < Cn; k0 += 32) {
    v16h a[4];
#pragma unroll
    for (int mi = 0; mi < 4; ++mi)
      a[mi] = load_afrag(X + (size_t)(m0 + 16 * mi) * Cn, Cn, k0);
#pragma unroll
    for (int f = 0; f < 4; ++f) {
      v16h bf = load_bfrag(W + (size_t)(n0 + 16 * f) * Cn, Cn, k0);
#pragma unroll
      for (int mi = 0; mi < 4; ++mi) acc[mi][f] = wmma16(a[mi], bf, acc[mi][f]);
    }
  }

#pragma unroll
  for (int mi = 0; mi < 4; ++mi)
#pragma unroll
    for (int f = 0; f < 4; ++f) {
      int col = n0 + lan + 16 * f;
      float bvv = bias[col];
#pragma unroll
      for (int i = 0; i < 8; ++i) {
        int r = m0 + mi * 16 + i + 8 * half;
        out[(size_t)r * Cn + col] = acc[mi][f][i] + bvv;
      }
    }
}

// ---------------------------------------------------------------- launcher
extern "C" void kernel_launch(void* const* d_in, const int* in_sizes, int n_in,
                              void* d_out, int out_size, void* d_ws, size_t ws_size,
                              hipStream_t stream) {
  const float* x  = (const float*)d_in[0];
  const float* Wq = (const float*)d_in[1];
  const float* bq = (const float*)d_in[2];
  const float* Wk = (const float*)d_in[3];
  const float* bk = (const float*)d_in[4];
  const float* Wv = (const float*)d_in[5];
  const float* bv = (const float*)d_in[6];
  const float* Wp = (const float*)d_in[7];
  const float* bp = (const float*)d_in[8];
  float* out = (float*)d_out;

  char* ws = (char*)d_ws;
  const size_t MB = 1u << 20;
  _Float16* xh  = (_Float16*)(ws);               //  8 MB  x as f16
  _Float16* Wqh = (_Float16*)(ws + 8 * MB);      //  2 MB
  _Float16* Wkh = (_Float16*)(ws + 10 * MB);     //  2 MB
  _Float16* Wvh = (_Float16*)(ws + 12 * MB);     //  2 MB
  _Float16* Wph = (_Float16*)(ws + 14 * MB);     //  2 MB
  _Float16* Qh  = (_Float16*)(ws + 16 * MB);     //  8 MB  [b,h,t,d]
  _Float16* Kh  = (_Float16*)(ws + 24 * MB);     //  8 MB  [b,h,t,d]
  _Float16* Vth = (_Float16*)(ws + 32 * MB);     //  8 MB  [b,h,d,t]
  _Float16* Yh  = (_Float16*)(ws + 40 * MB);     //  8 MB  [b*t, c]

  cvt_f32_f16_kernel<<<512, 256, 0, stream>>>(x,  xh,  (Bn * Tn * Cn) / 4);
  cvt_f32_f16_kernel<<<256, 256, 0, stream>>>(Wq, Wqh, (Cn * Cn) / 4);
  cvt_f32_f16_kernel<<<256, 256, 0, stream>>>(Wk, Wkh, (Cn * Cn) / 4);
  cvt_f32_f16_kernel<<<256, 256, 0, stream>>>(Wv, Wvh, (Cn * Cn) / 4);
  cvt_f32_f16_kernel<<<256, 256, 0, stream>>>(Wp, Wph, (Cn * Cn) / 4);

  // (B*T/64) * (C/64) = 64*16 = 1024 waves per matrix, 8 waves/block, y = {Q,K,V}
  qkv_gemm_kernel<<<dim3(128, 3), 256, 0, stream>>>(xh, Wqh, bq, Qh,
                                                    Wkh, bk, Kh,
                                                    Wvh, bv, Vth);

  // B*H * (T/32) = 32*64 = 2048 waves, 4 waves/block -> 512 blocks
  flash_attn_kernel<<<512, 128, 0, stream>>>(Qh, Kh, Vth, Yh);

  proj_gemm_kernel<<<128, 256, 0, stream>>>(Yh, Wph, bp, out);
}
